// FLabelingNet_41351945126314
// MI455X (gfx1250) — matrix operand order repaired and checked by reading, not compiled, not run on hardware
//
#include <hip/hip_runtime.h>

// ---------------------------------------------------------------------------
// FLabelingNet forward for MI455X (gfx1250, wave32, WMMA).
//
// Per layer:
//   x0  = x @ f0_w + f0_b          (50000x256x256 GEMM, WMMA f16->f32)
//   x0[idx] = x[idx] @ f1_w + f1_b (8192-row gather/scatter WMMA GEMM)
//   h   = x0 @ conv_w + conv_b     (50000x256x256 GEMM, WMMA f16->f32)
//   x'  = segment_sum(h[src], dst) (800k edges, hw global_atomic_add_f32;
//                                   h and x' both fit in the 192MB L2)
// Output: x[pos]  (4096,2,256) fp32.
//
// Precision: inputs are fp32 ~N(0,1); A/B operands converted to _Float16
// (better mantissa than bf16 at this scale, no overflow risk), accumulate f32
// via v_wmma_f32_16x16x32_f16. Weights are pre-transposed to W^T[n][k] so the
// per-lane B fragment is 32 contiguous bytes.
// ---------------------------------------------------------------------------

typedef __attribute__((ext_vector_type(16))) _Float16 v16h;
typedef __attribute__((ext_vector_type(8)))  _Float16 v8h;
typedef __attribute__((ext_vector_type(4)))  _Float16 v4h;
typedef __attribute__((ext_vector_type(8)))  float    v8f;

#define HIDDEN   256
#define NNODES   50000
#define NEDGES   800000
#define NPAIRS   4096
#define NLAYERS  3
#define NLBL     (2 * NPAIRS)          // 8192 labeled rows
#define WMAT     (HIDDEN * HIDDEN)     // 65536 elements per weight matrix

// --------------------------- fp32 -> f16 convert ---------------------------
__global__ void cvt_f32_to_f16(const float* __restrict__ src,
                               _Float16* __restrict__ dst, int n) {
  int base = (blockIdx.x * blockDim.x + threadIdx.x) * 4;
  if (base + 3 >= n) return;                    // n is a multiple of 4
  float4 v = *(const float4*)(src + base);
  v4h h = { (_Float16)v.x, (_Float16)v.y, (_Float16)v.z, (_Float16)v.w };
  *(v4h*)(dst + base) = h;
}

// ------------------- weight transpose + convert to f16 ---------------------
// wt layout: [9][n][k] with wt[m][n][k] = w_m[k][n];  m: 0-2 f0, 3-5 f1, 6-8 conv
__global__ void prep_weights(const float* __restrict__ w0,
                             const float* __restrict__ w1,
                             const float* __restrict__ w2,
                             _Float16* __restrict__ wt) {
  int t = blockIdx.x * blockDim.x + threadIdx.x;
  if (t >= 9 * WMAT) return;
  int m  = t >> 16;
  int nk = t & (WMAT - 1);
  int n  = nk >> 8;
  int k  = nk & 255;
  const float* src = (m < 3) ? (w0 + m * WMAT)
                   : (m < 6) ? (w1 + (m - 3) * WMAT)
                             : (w2 + (m - 6) * WMAT);
  wt[t] = (_Float16)src[k * HIDDEN + n];
}

// ------------------------------ WMMA GEMM ----------------------------------
// C[M x 256] = A[M x 256] @ W + bias.  A f16 row-major (via optional inMap row
// gather), W given as W^T[n][k] f16, out fp32 or f16 (via optional outMap row
// scatter). One wave computes one 16x16 tile; 8 waves per block.
__global__ __launch_bounds__(256)
void gemm_wmma_f16(const _Float16* __restrict__ A,
                   const _Float16* __restrict__ WT,
                   const float* __restrict__ bias,
                   void* __restrict__ out,
                   const int* __restrict__ inMap,
                   const int* __restrict__ outMap,
                   int M, int outIsF16) {
  const int wave    = threadIdx.x >> 5;
  const int lane    = threadIdx.x & 31;
  const int rowTile = blockIdx.x * 8 + wave;
  if (rowTile * 16 >= M) return;               // uniform per wave: EXEC stays all-1
  const int n0   = blockIdx.y * 16;
  const int half = lane >> 4;                  // 0: lanes 0-15, 1: lanes 16-31
  const int l15  = lane & 15;
  const int n    = n0 + l15;

  // A fragment source row for this lane (ISA 7.12.2: lane holds row M=lane&15)
  const int mA = rowTile * 16 + l15;
  const long rA = inMap ? (long)inMap[mA] : (long)mA;
  const _Float16* __restrict__ arow = A + rA * HIDDEN;
  const _Float16* __restrict__ brow = WT + (long)n * HIDDEN;

  // Bias folded into the accumulator: every D element of this lane has col n.
  const float bv = bias[n];
  v8f acc = { bv, bv, bv, bv, bv, bv, bv, bv };

#pragma unroll
  for (int k0 = 0; k0 < HIDDEN; k0 += 32) {
    // A 16x32 f16 frag: a[j]=A[m][k0+half*16+j]
    v16h a = *(const v16h*)(arow + k0 + half * 16);
    // B 32x16 frag (row-striped across lanes): b[j] = W[k0+half*16+j][n]
    //  = WT[n][k0+half*16+j] -> 32 contiguous bytes.
    v16h b = *(const v16h*)(brow + k0 + half * 16);

    acc = __builtin_amdgcn_wmma_f32_16x16x32_f16(
        /*neg_a=*/false, a, /*neg_b=*/false, b,
        /*c_mod=*/(short)0, acc, /*reuse_a=*/false, /*reuse_b=*/false);
  }

  // D layout: VGPR r -> row (half*8 + r), col l15.
  if (outIsF16) {
    _Float16* o = (_Float16*)out;
#pragma unroll
    for (int r = 0; r < 8; ++r) {
      int mo = rowTile * 16 + half * 8 + r;
      long ro = outMap ? (long)outMap[mo] : (long)mo;
      o[ro * HIDDEN + n] = (_Float16)acc[r];
    }
  } else {
    float* o = (float*)out;
#pragma unroll
    for (int r = 0; r < 8; ++r) {
      int mo = rowTile * 16 + half * 8 + r;
      long ro = outMap ? (long)outMap[mo] : (long)mo;
      o[ro * HIDDEN + n] = acc[r];
    }
  }
}

// -------------------- edge aggregation (segment_sum) -----------------------
// 64 threads per edge, 4 floats each; hardware f32 atomics (agent scope -> L2).
__global__ void edge_agg(const float* __restrict__ h,
                         const int* __restrict__ src,
                         const int* __restrict__ dst,
                         float* __restrict__ x) {
  long t = (long)blockIdx.x * blockDim.x + threadIdx.x;
  long e = t >> 6;
  if (e >= NEDGES) return;
  int c = (int)(t & 63) * 4;
  float4 v = *(const float4*)(h + (long)src[e] * HIDDEN + c);
  float* o = x + (long)dst[e] * HIDDEN + c;
  __hip_atomic_fetch_add(o + 0, v.x, __ATOMIC_RELAXED, __HIP_MEMORY_SCOPE_AGENT);
  __hip_atomic_fetch_add(o + 1, v.y, __ATOMIC_RELAXED, __HIP_MEMORY_SCOPE_AGENT);
  __hip_atomic_fetch_add(o + 2, v.z, __ATOMIC_RELAXED, __HIP_MEMORY_SCOPE_AGENT);
  __hip_atomic_fetch_add(o + 3, v.w, __ATOMIC_RELAXED, __HIP_MEMORY_SCOPE_AGENT);
}

// ------------------------- final gather x[pos] -----------------------------
__global__ void gather_out(const float* __restrict__ x,
                           const int* __restrict__ pos,
                           float* __restrict__ out) {
  long t = (long)blockIdx.x * blockDim.x + threadIdx.x;
  long i = t >> 6;
  if (i >= NLBL) return;
  int c = (int)(t & 63) * 4;
  *(float4*)(out + i * HIDDEN + c) =
      *(const float4*)(x + (long)pos[i] * HIDDEN + c);
}

// ---------------------------------------------------------------------------
extern "C" void kernel_launch(void* const* d_in, const int* in_sizes, int n_in,
                              void* d_out, int out_size, void* d_ws, size_t ws_size,
                              hipStream_t stream) {
  const float* x_in = (const float*)d_in[0];
  const float* f0w  = (const float*)d_in[1];
  const float* f0b  = (const float*)d_in[2];
  const float* f1w  = (const float*)d_in[3];
  const float* f1b  = (const float*)d_in[4];
  const float* cw   = (const float*)d_in[5];
  const float* cb   = (const float*)d_in[6];
  const int* esrc   = (const int*)d_in[7];
  const int* edst   = (const int*)d_in[8];
  const int* pos    = (const int*)d_in[9];   // (4096,2) flattened = idx

  // Workspace carve-up (~148 MB total):
  char* ws = (char*)d_ws;
  float*    X    = (float*)ws;    ws += (size_t)NNODES * HIDDEN * sizeof(float);
  float*    H    = (float*)ws;    ws += (size_t)NNODES * HIDDEN * sizeof(float);
  _Float16* xh   = (_Float16*)ws; ws += (size_t)NNODES * HIDDEN * sizeof(_Float16);
  _Float16* x0h  = (_Float16*)ws; ws += (size_t)NNODES * HIDDEN * sizeof(_Float16);
  _Float16* WT   = (_Float16*)ws; ws += (size_t)9 * WMAT * sizeof(_Float16);
  (void)ws_size; (void)in_sizes; (void)n_in; (void)out_size;

  // Convert + transpose all 9 weight matrices once.
  prep_weights<<<(9 * WMAT + 255) / 256, 256, 0, stream>>>(f0w, f1w, cw, WT);

  const int nel = NNODES * HIDDEN;               // 12.8M, multiple of 4
  const dim3 gFull((NNODES / 16 + 7) / 8, HIDDEN / 16);  // 3125 tiles, 16 col-tiles
  const dim3 gLbl((NLBL / 16 + 7) / 8, HIDDEN / 16);     // 512 tiles

  const float* xcur = x_in;
  for (int l = 0; l < NLAYERS; ++l) {
    // x (fp32) -> f16 operand buffer
    cvt_f32_to_f16<<<(nel / 4 + 255) / 256, 256, 0, stream>>>(xcur, xh, nel);

    // x0 = x @ f0_w + f0_b    (written directly as f16 for the conv GEMM)
    gemm_wmma_f16<<<gFull, 256, 0, stream>>>(
        xh, WT + (0 * 3 + l) * WMAT, f0b + l * HIDDEN, x0h,
        nullptr, nullptr, NNODES, /*outIsF16=*/1);

    // x0[idx] = x[idx] @ f1_w + f1_b   (gather rows, scatter rows; duplicate
    // idx entries write identical values so races are benign)
    gemm_wmma_f16<<<gLbl, 256, 0, stream>>>(
        xh, WT + (1 * 3 + l) * WMAT, f1b + l * HIDDEN, x0h,
        pos, pos, NLBL, /*outIsF16=*/1);

    // h = x0 @ conv_w + conv_b  (fp32 out)
    gemm_wmma_f16<<<gFull, 256, 0, stream>>>(
        x0h, WT + (2 * 3 + l) * WMAT, cb + l * HIDDEN, H,
        nullptr, nullptr, NNODES, /*outIsF16=*/0);

    // x' = segment_sum(h[src], dst)
    hipMemsetAsync(X, 0, (size_t)NNODES * HIDDEN * sizeof(float), stream);
    edge_agg<<<(unsigned)(((long)NEDGES * 64) / 256), 256, 0, stream>>>(
        H, esrc, edst, X);

    xcur = X;
  }

  // out = x[pos]  -> (4096, 2, 256) fp32
  gather_out<<<(NLBL * 64) / 256, 256, 0, stream>>>(X, pos, (float*)d_out);
}
